// MultiHeadCrossAttention_1812476198932
// MI455X (gfx1250) — matrix operand order repaired and checked by reading, not compiled, not run on hardware
//
#include <hip/hip_runtime.h>
#include <hip/hip_bf16.h>
#include <cstdint>

typedef _Float16 v16h __attribute__((ext_vector_type(16)));
typedef _Float16 v8h  __attribute__((ext_vector_type(8)));
typedef float    v8f  __attribute__((ext_vector_type(8)));

static __device__ __forceinline__ v8f wmma16x16x32(v16h a, v16h b, v8f c) {
  // D = A(16x32 f16) * B(32x16 f16) + C(16x16 f32)
  return __builtin_amdgcn_wmma_f32_16x16x32_f16(false, a, false, b, (short)0, c,
                                                false, false);
}

constexpr int   Bn = 8, T = 448, S = 1500, D = 1024, H = 16, Dh = 64;
constexpr int   SP = 1536;                 // padded S for V^T
constexpr int   BT = Bn * T;               // 3584
constexpr int   BS = Bn * S;               // 12000
constexpr int   TT = T / 16;               // 28 t-tiles
constexpr int   ST64 = (S + 63) / 64;      // 24 s-tiles of 64
constexpr int   ROWS = Bn * H * T;         // 57344 softmax rows
constexpr int   NT = D / 64;               // 16 n-tiles per GEMM
constexpr size_t OUT_ELEMS = (size_t)Bn * T * D;   // 3,670,016
constexpr float QK_SCALE = 0.35355339059327373f;   // 64^-0.25

// ---------------------------------------------------------------- conversions
__global__ void cvt_f32_to_f16(const float* __restrict__ in,
                               _Float16* __restrict__ out, int n) {
  int i = blockIdx.x * blockDim.x + threadIdx.x;
  if (i < n) out[i] = (_Float16)in[i];
}

// W [K=D, N=D] f32 row-major -> Wt [N, K] f16 (B-fragment friendly)
__global__ void transpose_w_f16(const float* __restrict__ W,
                                _Float16* __restrict__ Wt) {
  __shared__ float tile[32][33];
  int tx = threadIdx.x, ty = threadIdx.y;          // (32,8)
  int n0 = blockIdx.x * 32, k0 = blockIdx.y * 32;
#pragma unroll
  for (int i = 0; i < 4; i++)
    tile[ty + 8 * i][tx] = W[(size_t)(k0 + ty + 8 * i) * D + n0 + tx];
  __syncthreads();
#pragma unroll
  for (int i = 0; i < 4; i++)
    Wt[(size_t)(n0 + ty + 8 * i) * D + k0 + tx] = (_Float16)tile[tx][ty + 8 * i];
}

// v_h [B, S, D] f16 -> vt [B*D, SP] f16 (row = b*1024 + h*64 + dh), zero pad S..SP
__global__ void transpose_v_f16(const _Float16* __restrict__ vh,
                                _Float16* __restrict__ vt) {
  __shared__ _Float16 tile[32][33];
  int tx = threadIdx.x, ty = threadIdx.y;          // (32,8)
  int s0 = blockIdx.x * 32, d0 = blockIdx.y * 32, b = blockIdx.z;
#pragma unroll
  for (int i = 0; i < 4; i++) {
    int s = s0 + ty + 8 * i;
    _Float16 val = (_Float16)0.f;
    if (s < S) val = vh[((size_t)b * S + s) * D + d0 + tx];
    tile[ty + 8 * i][tx] = val;
  }
  __syncthreads();
#pragma unroll
  for (int i = 0; i < 4; i++)
    vt[((size_t)b * D + d0 + ty + 8 * i) * SP + s0 + tx] = tile[tx][ty + 8 * i];
}

// ----------------------------------------------------- GEMM with async LDS B
// C[M,D] = (A[M,D] @ Bt[D,D]^T (+bias)) * scale
// Block = 8 waves sharing one 64-wide N tile; each wave owns one 16-row M tile.
// B slab (64 N x 32 K, 4KB f16) staged to LDS with global_load_async_to_lds,
// double-buffered; per-wave s_wait_asynccnt + block barrier publishes slabs.
template <bool F16OUT, bool HASBIAS>
__global__ __launch_bounds__(256) void gemm_wmma_lds(
    const _Float16* __restrict__ A, const _Float16* __restrict__ Bt,
    const float* __restrict__ bias, float scale, void* __restrict__ Cout,
    int M) {
  __shared__ _Float16 bslab[2][64 * 32];           // 2 x 4KB
  int tid = threadIdx.x;
  int wave = tid >> 5, lane = tid & 31;
  int lane15 = lane & 15, hi = lane >> 4;
  int tilesM = M >> 4;
  int tile_n = blockIdx.x % NT;
  int tile_m = (blockIdx.x / NT) * 8 + wave;
  if (tile_m >= tilesM) tile_m = tilesM - 1;       // clamp: duplicate work, no
  int m0 = tile_m * 16, n0 = tile_n * 64;          // early exit (barriers!)

  // cooperative stage: thread tid copies 8 halves (16B) of the 64x32 slab
  int crow = tid >> 2;                             // 0..63 (local n)
  int cchunk = (tid & 3) * 8;                      // 0,8,16,24 (local k)
  const _Float16* gsrc = Bt + (size_t)(n0 + crow) * D + cchunk;
  uint32_t lds0 = (uint32_t)(uintptr_t)(&bslab[0][0]);
  uint32_t ldsoff = (uint32_t)((crow * 32 + cchunk) * sizeof(_Float16));

  auto issue_slab = [&](int buf, int k0) {
    uint64_t gaddr = (uint64_t)(uintptr_t)(gsrc + k0);
    uint32_t laddr = lds0 + (uint32_t)(buf * 64 * 32 * 2) + ldsoff;
    asm volatile("global_load_async_to_lds_b128 %0, %1, off"
                 :: "v"(laddr), "v"(gaddr)
                 : "memory");
  };

  issue_slab(0, 0);
  const _Float16* arow = A + (size_t)(m0 + lane15) * D;
  v8f acc[4] = {};
  asm volatile("s_wait_asynccnt 0x0" ::: "memory");
  __syncthreads();

  for (int k0 = 0, it = 0; k0 < D; k0 += 32, ++it) {
    int cur = it & 1;
    if (k0 + 32 < D) issue_slab(cur ^ 1, k0 + 32);
    v8h alo = *(const v8h*)(arow + k0 + hi * 8);
    v8h ahi = *(const v8h*)(arow + k0 + 16 + hi * 8);
    v16h a;
#pragma unroll
    for (int i = 0; i < 8; i++) { a[i] = alo[i]; a[8 + i] = ahi[i]; }
#pragma unroll
    for (int t = 0; t < 4; t++) {
      const _Float16* bp = &bslab[cur][(t * 16 + lane15) * 32 + hi * 8];
      v8h blo = *(const v8h*)bp;
      v8h bhi = *(const v8h*)(bp + 16);
      v16h bb;
#pragma unroll
      for (int i = 0; i < 8; i++) { bb[i] = blo[i]; bb[8 + i] = bhi[i]; }
      acc[t] = wmma16x16x32(a, bb, acc[t]);
    }
    asm volatile("s_wait_asynccnt 0x0" ::: "memory");
    __syncthreads();   // next slab landed for everyone; cur slab fully drained
  }

#pragma unroll
  for (int t = 0; t < 4; t++) {
    int n = n0 + t * 16 + lane15;
    float bvl = HASBIAS ? bias[n] : 0.0f;
#pragma unroll
    for (int r = 0; r < 8; r++) {
      int row = m0 + r + hi * 8;
      float val = (acc[t][r] + bvl) * scale;
      if (F16OUT)
        ((_Float16*)Cout)[(size_t)row * D + n] = (_Float16)val;
      else
        ((float*)Cout)[(size_t)row * D + n] = val;
    }
  }
}

// ---------------------------------------------------------- QK^T (logits)
// qk[b,h,t,s] = sum_d q[b,t,h,d]*k[b,s,h,d]  (scale pre-applied to q,k)
__global__ __launch_bounds__(256) void qk_wmma_kernel(
    const _Float16* __restrict__ qh, const _Float16* __restrict__ kh,
    float* __restrict__ qk) {
  int tid = blockIdx.x * blockDim.x + threadIdx.x;
  int wave = tid >> 5, lane = tid & 31;
  int lane15 = lane & 15, hi = lane >> 4;
  int sTile = wave % ST64;
  int tTile = (wave / ST64) % TT;
  int bh = wave / (ST64 * TT);
  if (bh >= Bn * H) return;
  int b = bh >> 4, h = bh & 15;
  int t0 = tTile * 16, s0 = sTile * 64;
  const _Float16* qrow = qh + (size_t)(b * T + t0 + lane15) * D + h * Dh;
  v8f acc[4] = {};
#pragma unroll
  for (int k0 = 0; k0 < Dh; k0 += 32) {
    v8h alo = *(const v8h*)(qrow + k0 + hi * 8);
    v8h ahi = *(const v8h*)(qrow + k0 + 16 + hi * 8);
    v16h a;
#pragma unroll
    for (int i = 0; i < 8; i++) { a[i] = alo[i]; a[8 + i] = ahi[i]; }
#pragma unroll
    for (int t = 0; t < 4; t++) {
      int s = s0 + t * 16 + lane15;
      if (s > S - 1) s = S - 1;  // clamp loads; stores guarded below
      const _Float16* krow =
          kh + (size_t)(b * S + s) * D + h * Dh + k0 + hi * 8;
      v8h blo = *(const v8h*)krow;
      v8h bhi = *(const v8h*)(krow + 16);
      v16h bb;
#pragma unroll
      for (int i = 0; i < 8; i++) { bb[i] = blo[i]; bb[8 + i] = bhi[i]; }
      acc[t] = wmma16x16x32(a, bb, acc[t]);
    }
  }
  float* orow = qk + (size_t)bh * T * S;
#pragma unroll
  for (int t = 0; t < 4; t++) {
    int s = s0 + t * 16 + lane15;
    if (s < S) {
#pragma unroll
      for (int r = 0; r < 8; r++)
        orow[(size_t)(t0 + r + hi * 8) * S + s] = acc[t][r];
    }
  }
}

// --------------------------------------------------- softmax row statistics
__global__ __launch_bounds__(256) void softmax_stats_kernel(
    const float* __restrict__ qk, float* __restrict__ rmax,
    float* __restrict__ rinv) {
  int tid = blockIdx.x * blockDim.x + threadIdx.x;
  int wave = tid >> 5, lane = tid & 31;
  if (wave >= ROWS) return;
  const float* row = qk + (size_t)wave * S;
  float m = -3.0e38f;
  for (int s = lane; s < S; s += 32) m = fmaxf(m, row[s]);
#pragma unroll
  for (int off = 16; off; off >>= 1) m = fmaxf(m, __shfl_xor(m, off, 32));
  float sum = 0.f;
  for (int s = lane; s < S; s += 32) sum += __expf(row[s] - m);
#pragma unroll
  for (int off = 16; off; off >>= 1) sum += __shfl_xor(sum, off, 32);
  if (lane == 0) {
    rmax[wave] = m;
    rinv[wave] = 1.0f / sum;
  }
}

// ------------------------------------------- softmax(P) @ V, fused on the fly
__global__ __launch_bounds__(256) void wv_wmma_kernel(
    const float* __restrict__ qk, const float* __restrict__ rmax,
    const float* __restrict__ rinv, const _Float16* __restrict__ vt,
    _Float16* __restrict__ wvh) {
  int tid = blockIdx.x * blockDim.x + threadIdx.x;
  int wave = tid >> 5, lane = tid & 31;
  int lane15 = lane & 15, hi = lane >> 4;
  int tTile = wave % TT;
  int bh = wave / TT;
  if (bh >= Bn * H) return;
  int b = bh >> 4, h = bh & 15;
  int t0 = tTile * 16;
  int row = bh * T + t0 + lane15;
  float rm = rmax[row], ri = rinv[row];
  const float* L = qk + (size_t)row * S;
  v8f acc[4] = {};
  for (int s0 = 0; s0 < S; s0 += 32) {
    v16h a;
    if (s0 + 32 <= S) {  // fast path: fully in-bounds 32-wide K slab
      float4 l0 = *(const float4*)(L + s0 + hi * 8);
      float4 l1 = *(const float4*)(L + s0 + hi * 8 + 4);
      float4 l2 = *(const float4*)(L + s0 + 16 + hi * 8);
      float4 l3 = *(const float4*)(L + s0 + 16 + hi * 8 + 4);
      float lv[16] = {l0.x, l0.y, l0.z, l0.w, l1.x, l1.y, l1.z, l1.w,
                      l2.x, l2.y, l2.z, l2.w, l3.x, l3.y, l3.z, l3.w};
#pragma unroll
      for (int e = 0; e < 16; e++)
        a[e] = (_Float16)(__expf(lv[e] - rm) * ri);
    } else {  // tail (s0 = 1472): zero probabilities for s >= S
#pragma unroll
      for (int e = 0; e < 16; e++) {
        int s = s0 + (e >> 3) * 16 + hi * 8 + (e & 7);
        float p = 0.f;
        if (s < S) p = __expf(L[s] - rm) * ri;
        a[e] = (_Float16)p;
      }
    }
#pragma unroll
    for (int t = 0; t < 4; t++) {
      const _Float16* vr =
          vt + ((size_t)b * D + h * Dh + t * 16 + lane15) * SP + s0 + hi * 8;
      v8h blo = *(const v8h*)vr;
      v8h bhi = *(const v8h*)(vr + 16);
      v16h bb;
#pragma unroll
      for (int i = 0; i < 8; i++) { bb[i] = blo[i]; bb[8 + i] = bhi[i]; }
      acc[t] = wmma16x16x32(a, bb, acc[t]);
    }
  }
#pragma unroll
  for (int t = 0; t < 4; t++)
#pragma unroll
    for (int r = 0; r < 8; r++)
      wvh[(size_t)(b * T + t0 + r + hi * 8) * D + h * Dh + t * 16 + lane15] =
          (_Float16)acc[t][r];
}

// ---------------------------------------------------------------- dispatcher
extern "C" void kernel_launch(void* const* d_in, const int* in_sizes, int n_in,
                              void* d_out, int out_size, void* d_ws,
                              size_t ws_size, hipStream_t stream) {
  (void)in_sizes; (void)n_in; (void)out_size; (void)ws_size;
  const float* x  = (const float*)d_in[0];
  const float* xa = (const float*)d_in[1];
  const float* Wq = (const float*)d_in[2];
  const float* bq = (const float*)d_in[3];
  const float* Wk = (const float*)d_in[4];
  const float* Wv = (const float*)d_in[5];
  const float* bv = (const float*)d_in[6];
  const float* Wo = (const float*)d_in[7];
  const float* bo = (const float*)d_in[8];
  float* out = (float*)d_out;
  float* qk  = out + OUT_ELEMS;  // tuple (out, qk) concatenated

  _Float16* p = (_Float16*)d_ws;
  _Float16* x_h  = p; p += (size_t)BT * D;
  _Float16* xa_h = p; p += (size_t)BS * D;
  _Float16* wqt  = p; p += (size_t)D * D;
  _Float16* wkt  = p; p += (size_t)D * D;
  _Float16* wvt  = p; p += (size_t)D * D;
  _Float16* wot  = p; p += (size_t)D * D;
  _Float16* q_h  = p; p += (size_t)BT * D;
  _Float16* k_h  = p; p += (size_t)BS * D;
  _Float16* v_h  = p; p += (size_t)BS * D;
  _Float16* vtb  = p; p += (size_t)Bn * D * SP;
  _Float16* wv_h = p; p += (size_t)BT * D;
  float* rmaxA = (float*)p;
  float* rinvA = rmaxA + ROWS;

  dim3 b256(256);
  cvt_f32_to_f16<<<(BT * D + 255) / 256, b256, 0, stream>>>(x, x_h, BT * D);
  cvt_f32_to_f16<<<(BS * D + 255) / 256, b256, 0, stream>>>(xa, xa_h, BS * D);

  dim3 tb(32, 8), tg(D / 32, D / 32);
  transpose_w_f16<<<tg, tb, 0, stream>>>(Wq, wqt);
  transpose_w_f16<<<tg, tb, 0, stream>>>(Wk, wkt);
  transpose_w_f16<<<tg, tb, 0, stream>>>(Wv, wvt);
  transpose_w_f16<<<tg, tb, 0, stream>>>(Wo, wot);

  // projections: q = (x Wq + bq)*s ; k = (xa Wk)*s ; v = xa Wv + bv
  int gBT = ((BT / 16 + 7) / 8) * NT;   // 28*16 = 448 blocks
  int gBS = ((BS / 16 + 7) / 8) * NT;   // 94*16 = 1504 blocks
  gemm_wmma_lds<true, true ><<<gBT, b256, 0, stream>>>(x_h,  wqt, bq, QK_SCALE, q_h, BT);
  gemm_wmma_lds<true, false><<<gBS, b256, 0, stream>>>(xa_h, wkt, nullptr, QK_SCALE, k_h, BS);
  gemm_wmma_lds<true, true ><<<gBS, b256, 0, stream>>>(xa_h, wvt, bv, 1.0f, v_h, BS);

  transpose_v_f16<<<dim3(SP / 32, D / 32, Bn), tb, 0, stream>>>(v_h, vtb);

  {  // attention logits -> d_out qk region
    int waves = Bn * H * TT * ST64;  // 86016
    qk_wmma_kernel<<<(waves * 32) / 256, b256, 0, stream>>>(q_h, k_h, qk);
  }
  softmax_stats_kernel<<<(ROWS * 32) / 256, b256, 0, stream>>>(qk, rmaxA, rinvA);
  {  // softmax @ V
    int waves = Bn * H * TT;  // 3584
    wv_wmma_kernel<<<(waves * 32) / 256, b256, 0, stream>>>(qk, rmaxA, rinvA, vtb, wv_h);
  }
  // final projection: out = wv Wo + bo (f32)
  gemm_wmma_lds<false, true><<<gBT, b256, 0, stream>>>(wv_h, wot, bo, 1.0f, out, BT);
}
